// MultiHeadAttention_24077586661974
// MI455X (gfx1250) — compile-verified, matrix operand order checked by reading, not run on hardware
//
#include <hip/hip_runtime.h>

// ---------------------------------------------------------------------------
// Problem constants (from reference)
// ---------------------------------------------------------------------------
constexpr int BATCH  = 4;
constexpr int SEQ    = 2048;
constexpr int DMODEL = 2048;
constexpr int NHEADS = 16;
constexpr int HDIM   = 128;          // head dim
constexpr int NQKV   = 3 * DMODEL;   // 6144
constexpr int MROWS  = BATCH * SEQ;  // 8192

// ---------------------------------------------------------------------------
// CDNA5 WMMA types (wave32): A/B = 16 x bf16, C/D = 8 x f32
// ---------------------------------------------------------------------------
typedef __attribute__((ext_vector_type(16))) __bf16 v16bf;
typedef __attribute__((ext_vector_type(8)))  float  v8f;

union FragBF {           // 32 bytes = 16 bf16 = one A or B fragment
    uint4 u[2];
    v16bf v;
};

// ---- f32 -> bf16 conversion: prefer the single-instruction packed cvt -----
#if __has_builtin(__builtin_amdgcn_cvt_pk_bf16_f32)
__device__ __forceinline__ unsigned pack2bf(float lo, float hi) {
    auto r = __builtin_amdgcn_cvt_pk_bf16_f32(lo, hi);   // v_cvt_pk_bf16_f32
    return __builtin_bit_cast(unsigned, r);
}
#else
__device__ __forceinline__ unsigned pack2bf_one(float f) {
    unsigned u = __builtin_bit_cast(unsigned, f);
    u = u + 0x7FFFu + ((u >> 16) & 1u);                  // RNE
    return u >> 16;
}
__device__ __forceinline__ unsigned pack2bf(float lo, float hi) {
    return pack2bf_one(lo) | (pack2bf_one(hi) << 16);
}
#endif
__device__ __forceinline__ unsigned short f2bf(float f) {
    return (unsigned short)pack2bf(f, f);
}
__device__ __forceinline__ float bf2f(unsigned short h) {
    unsigned u = ((unsigned)h) << 16;
    return __builtin_bit_cast(float, u);
}

__device__ __forceinline__ v8f wmma_bf16(const v16bf& a, const v16bf& b, const v8f& c) {
    return __builtin_amdgcn_wmma_f32_16x16x32_bf16(false, a, false, b, (short)0, c, false, false);
}

// 16-lane row reductions (row lives in one 16-lane half of a wave32)
__device__ __forceinline__ float rowmax16(float v) {
    for (int o = 1; o < 16; o <<= 1) v = fmaxf(v, __shfl_xor(v, o, 32));
    return v;
}
__device__ __forceinline__ float rowsum16(float v) {
    for (int o = 1; o < 16; o <<= 1) v += __shfl_xor(v, o, 32);
    return v;
}

// ---------------------------------------------------------------------------
// CDNA5 async global->LDS copies (ASYNCcnt), with synchronous fallback.
// Builtin signature (from toolchain diagnostic): param1 is
//   int __vector(4) addrspace(1)* ; param2 LDS side; then i32 offset, i32 cpol.
// ---------------------------------------------------------------------------
#if defined(__AMDGCN__) && __has_builtin(__builtin_amdgcn_global_load_async_to_lds_b128)
#define HAVE_ASYNC 1
#else
#define HAVE_ASYNC 0
#endif

typedef int v4i __attribute__((vector_size(16)));
typedef __attribute__((address_space(1))) v4i* gas_v4i;
typedef __attribute__((address_space(3))) v4i* las_v4i;

__device__ __forceinline__ void copy_b128_to_lds(const unsigned short* g, unsigned short* l) {
#if HAVE_ASYNC
    __builtin_amdgcn_global_load_async_to_lds_b128(
        (gas_v4i)(unsigned short*)g, (las_v4i)l, 0, 0);
#else
    *(uint4*)l = *(const uint4*)g;
#endif
}

template <int N>
__device__ __forceinline__ void async_wait() {
#if HAVE_ASYNC
#if __has_builtin(__builtin_amdgcn_s_wait_asynccnt)
    __builtin_amdgcn_s_wait_asynccnt(N);
#else
    asm volatile("s_wait_asynccnt %0" :: "i"(N) : "memory");
#endif
#endif
}

// ---------------------------------------------------------------------------
// Preprocessing: one-time f32 -> bf16 conversions (so GEMM staging is pure
// byte movement and every element is converted exactly once).
// ---------------------------------------------------------------------------
__global__ __launch_bounds__(256)
void pack_bf16_kernel(const float* __restrict__ in, unsigned short* __restrict__ out, int n4)
{
    const int i = blockIdx.x * 256 + threadIdx.x;
    if (i >= n4) return;
    float4 f = ((const float4*)in)[i];
    uint2 p;
    p.x = pack2bf(f.x, f.y);
    p.y = pack2bf(f.z, f.w);
    ((uint2*)out)[i] = p;
}

// W[K,N] f32 -> Wt[N,K] bf16 (tiled 32x32 via LDS; coalesced both sides)
__global__ __launch_bounds__(256)
void transpose_bf16_kernel(const float* __restrict__ in, unsigned short* __restrict__ out,
                           int K, int N)
{
    __shared__ float tile[32][33];
    const int t  = threadIdx.x;
    const int k0 = blockIdx.y * 32;
    const int n0 = blockIdx.x * 32;

    const int tx = t & 31, ty = t >> 5;                 // load: 32 wide x 8 rows
    for (int rr = 0; rr < 4; ++rr)
        tile[ty + rr * 8][tx] = in[(size_t)(k0 + ty + rr * 8) * N + n0 + tx];
    __syncthreads();

    const int dcol = t & 15, drow = t >> 4;             // store: dword = 2 k's
    for (int rr = 0; rr < 2; ++rr) {
        const int n = drow + rr * 16;
        *(unsigned*)&out[(size_t)(n0 + n) * K + k0 + 2 * dcol] =
            pack2bf(tile[2 * dcol][n], tile[2 * dcol + 1][n]);
    }
}

// ---------------------------------------------------------------------------
// GEMM: C[M,N] = A[M,K](bf16) @ Wt[N,K](bf16)^T + bias, via bf16 WMMA.
// Block = 256 thr (8 waves) computes 128x128. Per 32-deep K-step both tiles
// are 128 rows x 64B, double-buffered in LDS and filled with async b128
// global->LDS copies (4 per wave per tile): wait own ASYNCcnt<=4, barrier,
// compute, barrier, refill freed buffer. Padded 80B LDS rows keep the
// ds_load_b128 fragment reads bank-conflict-free.
// ---------------------------------------------------------------------------
constexpr int LDS_STRIDE = 40;   // 32 data + 8 pad (bf16 elements)

__device__ __forceinline__ void issue_tile(const unsigned short* Ag, const unsigned short* Bg,
                                           unsigned short* la, unsigned short* lb,
                                           int t, int K)
{
    for (int cp = 0; cp < 2; ++cp) {
        const int u = t + cp * 256;
        const int row = u >> 2, seg = (u & 3) * 8;
        copy_b128_to_lds(Ag + (size_t)row * K + seg, la + row * LDS_STRIDE + seg);
    }
    for (int cp = 0; cp < 2; ++cp) {
        const int u = t + cp * 256;
        const int row = u >> 2, seg = (u & 3) * 8;
        copy_b128_to_lds(Bg + (size_t)row * K + seg, lb + row * LDS_STRIDE + seg);
    }
}

template <bool OUT_BF16>
__global__ __launch_bounds__(256)
void gemm_wmma_bf16(const unsigned short* __restrict__ A, const unsigned short* __restrict__ Wt,
                    const float* __restrict__ bias, void* __restrict__ Cout,
                    int M, int N, int K)
{
    __shared__ unsigned short lds_a[2][128 * LDS_STRIDE];
    __shared__ unsigned short lds_b[2][128 * LDS_STRIDE];

    const int t    = threadIdx.x;
    const int wave = t >> 5;
    const int lane = t & 31;
    const int half = lane >> 4;          // 0: lanes 0-15, 1: lanes 16-31
    const int lm   = lane & 15;
    const int soff = half * 8;           // per-half K start (ISA 7.12.2)

    const int gm0 = blockIdx.y * 128;
    const int gn0 = blockIdx.x * 128;
    const int wm  = (wave >> 1) * 32;    // 4 waves along M (32 rows each)
    const int wn  = (wave & 1) * 64;     // 2 waves along N (64 cols each)

    v8f acc[2][4];
    for (int mt = 0; mt < 2; ++mt)
        for (int nt = 0; nt < 4; ++nt)
            for (int r = 0; r < 8; ++r) acc[mt][nt][r] = 0.f;

    const unsigned short* Abase = A  + (size_t)gm0 * K;
    const unsigned short* Bbase = Wt + (size_t)gn0 * K;
    const int T = K / 32;

    issue_tile(Abase, Bbase, lds_a[0], lds_b[0], t, K);
    if (T > 1) issue_tile(Abase + 32, Bbase + 32, lds_a[1], lds_b[1], t, K);

    for (int i = 0; i < T; ++i) {
        unsigned short* la = lds_a[i & 1];
        unsigned short* lb = lds_b[i & 1];

        if (i + 1 < T) async_wait<4>();  // my tile-i copies done; tile-i+1 may fly
        else           async_wait<0>();
        __syncthreads();                 // everyone's copies to this buffer done

        // ---- fragment loads (2x b128 each: k in [soff,+8) / [soff+16,+24))
        FragBF af[2], bfg[4];
        for (int mt = 0; mt < 2; ++mt) {
            const int m = wm + mt * 16 + lm;
            af[mt].u[0] = *(const uint4*)&la[m * LDS_STRIDE + soff];
            af[mt].u[1] = *(const uint4*)&la[m * LDS_STRIDE + soff + 16];
        }
        for (int nt = 0; nt < 4; ++nt) {
            const int n = wn + nt * 16 + lm;
            bfg[nt].u[0] = *(const uint4*)&lb[n * LDS_STRIDE + soff];
            bfg[nt].u[1] = *(const uint4*)&lb[n * LDS_STRIDE + soff + 16];
        }
        for (int mt = 0; mt < 2; ++mt)
            for (int nt = 0; nt < 4; ++nt)
                acc[mt][nt] = wmma_bf16(af[mt].v, bfg[nt].v, acc[mt][nt]);

        __syncthreads();                 // all waves finished reading buffer
        if (i + 2 < T)                   // refill the freed buffer
            issue_tile(Abase + (i + 2) * 32, Bbase + (i + 2) * 32, la, lb, t, K);
    }

    // ---- epilogue: C layout — vgpr r: row r (+8 for hi half), col = lm
    for (int mt = 0; mt < 2; ++mt)
        for (int nt = 0; nt < 4; ++nt) {
            const int n = gn0 + wn + nt * 16 + lm;
            const float bv = bias[n];
            for (int r = 0; r < 8; ++r) {
                const int m = gm0 + wm + mt * 16 + r + half * 8;
                const float v = acc[mt][nt][r] + bv;
                const size_t idx = (size_t)m * N + n;
                if constexpr (OUT_BF16) ((unsigned short*)Cout)[idx] = f2bf(v);
                else                    ((float*)Cout)[idx] = v;
            }
        }
}

// ---------------------------------------------------------------------------
// RoPE + head split + v transpose.
//   q,k -> [B,H,S,128] bf16 ; v -> [B,H,128,S] bf16 (PV B-frags contiguous)
// ---------------------------------------------------------------------------
__global__ __launch_bounds__(256)
void rope_split_kernel(const unsigned short* __restrict__ qkv,
                       const float* __restrict__ cosb, const float* __restrict__ sinb,
                       unsigned short* __restrict__ qo,
                       unsigned short* __restrict__ ko,
                       unsigned short* __restrict__ vto)
{
    const int t = blockIdx.x * 256 + threadIdx.x;   // B*H*S*64 total
    const int i = t & 63;
    const int s = (t >> 6) & (SEQ - 1);
    const int h = (t >> 17) & (NHEADS - 1);
    const int b = t >> 21;

    const size_t row = (size_t)(b * SEQ + s) * NQKV;
    const int col = h * HDIM + 2 * i;

    const unsigned qp = *(const unsigned*)&qkv[row + col];
    const unsigned kp = *(const unsigned*)&qkv[row + DMODEL + col];
    const unsigned vp = *(const unsigned*)&qkv[row + 2 * DMODEL + col];

    const float qe = bf2f((unsigned short)qp), qd = bf2f((unsigned short)(qp >> 16));
    const float ke = bf2f((unsigned short)kp), kd = bf2f((unsigned short)(kp >> 16));
    const float ve = bf2f((unsigned short)vp), vd = bf2f((unsigned short)(vp >> 16));

    const float c  = cosb[s * 64 + i];
    const float sn = sinb[s * 64 + i];

    const size_t qkidx = ((size_t)(b * NHEADS + h) * SEQ + s) * HDIM + 2 * i;
    *(unsigned*)&qo[qkidx] = pack2bf(qe * c - qd * sn, qe * sn + qd * c);
    *(unsigned*)&ko[qkidx] = pack2bf(ke * c - kd * sn, ke * sn + kd * c);

    const size_t vti = ((size_t)(b * NHEADS + h) * HDIM + 2 * i) * SEQ + s;
    const unsigned vv = pack2bf(ve, vd);
    vto[vti]       = (unsigned short)vv;
    vto[vti + SEQ] = (unsigned short)(vv >> 16);
}

// ---------------------------------------------------------------------------
// Causal flash attention. One wave per (b,h, 16-query tile).
// ---------------------------------------------------------------------------
constexpr float ATTN_SCALE = 0.0883883476483184f;   // 1/sqrt(128)
constexpr float NEG_BIG    = -3.0e38f;

__global__ __launch_bounds__(256)
void attn_wmma_kernel(const unsigned short* __restrict__ qm,
                      const unsigned short* __restrict__ km,
                      const unsigned short* __restrict__ vtm,
                      unsigned short* __restrict__ o)
{
    __shared__ unsigned short lds_p[8][16 * LDS_STRIDE];  // per-wave P tile

    const int t    = threadIdx.x;
    const int wave = t >> 5;
    const int lane = t & 31;
    const int half = lane >> 4;
    const int lm   = lane & 15;
    const int soff = half * 8;

    const int task = blockIdx.x * 8 + wave;   // 0 .. B*H*(SEQ/16)-1
    const int qt   = task & (SEQ / 16 - 1);   // query tile
    const int bh   = task >> 7;               // b*NHEADS + h
    const int q0   = qt * 16;

    const size_t qk_base = (size_t)bh * SEQ * HDIM;
    const size_t vt_base = (size_t)bh * HDIM * SEQ;

    FragBF qf[4];
    for (int c = 0; c < 4; ++c) {
        const unsigned short* p = qm + qk_base + (size_t)(q0 + lm) * HDIM + c * 32 + soff;
        qf[c].u[0] = *(const uint4*)p;
        qf[c].u[1] = *(const uint4*)(p + 16);
    }

    v8f oacc[8];
    for (int nt = 0; nt < 8; ++nt)
        for (int r = 0; r < 8; ++r) oacc[nt][r] = 0.f;
    float mi[8], li[8];
    for (int r = 0; r < 8; ++r) { mi[r] = NEG_BIG; li[r] = 0.f; }

    unsigned short* pbuf = lds_p[wave];
    const int nkb = (q0 + 15) / 32 + 1;       // causal: key blocks of 32

    for (int kb = 0; kb < nkb; ++kb) {
        const int kbase = kb * 32;

        if (kb + 1 < nkb) {                   // global_prefetch_b8 next block
            const unsigned short* nk = km + qk_base + (size_t)(kbase + 32 + lm) * HDIM;
            __builtin_prefetch(nk, 0, 3);
            __builtin_prefetch(nk + 16 * HDIM, 0, 3);
            __builtin_prefetch(vtm + vt_base + (size_t)lm * SEQ + kbase + 32, 0, 3);
        }

        v8f s0, s1;
        for (int r = 0; r < 8; ++r) { s0[r] = 0.f; s1[r] = 0.f; }
        for (int c = 0; c < 4; ++c) {
            FragBF k0f, k1f;
            const unsigned short* pk = km + qk_base + (size_t)(kbase + lm) * HDIM + c * 32 + soff;
            k0f.u[0] = *(const uint4*)pk;
            k0f.u[1] = *(const uint4*)(pk + 16);
            const unsigned short* pk1 = pk + 16 * HDIM;
            k1f.u[0] = *(const uint4*)pk1;
            k1f.u[1] = *(const uint4*)(pk1 + 16);
            s0 = wmma_bf16(qf[c].v, k0f.v, s0);
            s1 = wmma_bf16(qf[c].v, k1f.v, s1);
        }

        const bool diag = (kb == nkb - 1);
        for (int r = 0; r < 8; ++r) {
            float a = s0[r] * ATTN_SCALE;
            float b = s1[r] * ATTN_SCALE;
            if (diag) {
                const int m = q0 + r + half * 8;
                if (kbase + lm > m)      a = NEG_BIG;
                if (kbase + 16 + lm > m) b = NEG_BIG;
            }
            s0[r] = a; s1[r] = b;
        }

        float alpha[8];
        for (int r = 0; r < 8; ++r) {
            const float tmax = rowmax16(fmaxf(s0[r], s1[r]));
            const float mnew = fmaxf(mi[r], tmax);
            alpha[r] = __expf(mi[r] - mnew);
            mi[r] = mnew;
            const float p0 = __expf(s0[r] - mnew);
            const float p1 = __expf(s1[r] - mnew);
            s0[r] = p0; s1[r] = p1;
            li[r] = li[r] * alpha[r] + rowsum16(p0 + p1);
        }
        for (int nt = 0; nt < 8; ++nt)
            for (int r = 0; r < 8; ++r) oacc[nt][r] *= alpha[r];

        for (int r = 0; r < 8; ++r) {
            const int row = r + half * 8;
            const unsigned pp = pack2bf(s0[r], s1[r]);
            pbuf[row * LDS_STRIDE + lm]      = (unsigned short)pp;
            pbuf[row * LDS_STRIDE + 16 + lm] = (unsigned short)(pp >> 16);
        }
        FragBF pf;
        pf.u[0] = *(const uint4*)&pbuf[lm * LDS_STRIDE + soff];
        pf.u[1] = *(const uint4*)&pbuf[lm * LDS_STRIDE + soff + 16];

        for (int nt = 0; nt < 8; ++nt) {
            FragBF vf;
            const unsigned short* pv = vtm + vt_base + (size_t)(nt * 16 + lm) * SEQ + kbase + soff;
            vf.u[0] = *(const uint4*)pv;
            vf.u[1] = *(const uint4*)(pv + 16);
            oacc[nt] = wmma_bf16(pf.v, vf.v, oacc[nt]);
        }
    }

    // ---- normalize + store bf16 to [B, S, H*D] (A operand of out-proj GEMM)
    const int b = bh >> 4, h = bh & 15;
    for (int nt = 0; nt < 8; ++nt)
        for (int r = 0; r < 8; ++r) {
            const int row = q0 + r + half * 8;
            o[(size_t)(b * SEQ + row) * DMODEL + h * HDIM + nt * 16 + lm] =
                f2bf(oacc[nt][r] / li[r]);
        }
}

// ---------------------------------------------------------------------------
// Host-side launcher
// ---------------------------------------------------------------------------
extern "C" void kernel_launch(void* const* d_in, const int* in_sizes, int n_in,
                              void* d_out, int out_size, void* d_ws, size_t ws_size,
                              hipStream_t stream)
{
    (void)in_sizes; (void)n_in; (void)out_size; (void)ws_size;

    const float* x      = (const float*)d_in[0];
    const float* w_qkv  = (const float*)d_in[1];
    const float* b_qkv  = (const float*)d_in[2];
    const float* w_out  = (const float*)d_in[3];
    const float* b_out  = (const float*)d_in[4];
    const float* r_cos  = (const float*)d_in[5];
    const float* r_sin  = (const float*)d_in[6];

    // Workspace layout (bytes):
    //   x_bf16 [M,K]     : 32 MB   (aliased by o_bf16 after GEMM1)
    //   wt_qkv [6144,2048] bf16 : 24 MB
    //   wt_out [2048,2048] bf16 :  8 MB
    //   qkv    [M,6144] bf16    : 96 MB
    //   q,k    [B,H,S,128] bf16 : 32 MB each
    //   vT     [B,H,128,S] bf16 : 32 MB
    char* ws = (char*)d_ws;
    const size_t SZ_X    = (size_t)MROWS * DMODEL * 2;
    const size_t SZ_WQKV = (size_t)NQKV * DMODEL * 2;
    const size_t SZ_WOUT = (size_t)DMODEL * DMODEL * 2;
    const size_t SZ_QKV  = (size_t)MROWS * NQKV * 2;
    const size_t SZ_HEAD = (size_t)BATCH * NHEADS * SEQ * HDIM * 2;

    unsigned short* xb   = (unsigned short*)(ws);
    unsigned short* ob   = (unsigned short*)(ws);   // alias: x dead after GEMM1
    unsigned short* wtq  = (unsigned short*)(ws + SZ_X);
    unsigned short* wto  = (unsigned short*)(ws + SZ_X + SZ_WQKV);
    unsigned short* qkv  = (unsigned short*)(ws + SZ_X + SZ_WQKV + SZ_WOUT);
    unsigned short* qb   = (unsigned short*)(ws + SZ_X + SZ_WQKV + SZ_WOUT + SZ_QKV);
    unsigned short* kb   = (unsigned short*)(ws + SZ_X + SZ_WQKV + SZ_WOUT + SZ_QKV + SZ_HEAD);
    unsigned short* vtb  = (unsigned short*)(ws + SZ_X + SZ_WQKV + SZ_WOUT + SZ_QKV + 2 * SZ_HEAD);

    // 0) one-time bf16 conversion / weight transpose
    {
        const int n4 = MROWS * DMODEL / 4;
        pack_bf16_kernel<<<(n4 + 255) / 256, 256, 0, stream>>>(x, xb, n4);
        dim3 g1(NQKV / 32, DMODEL / 32);
        transpose_bf16_kernel<<<g1, 256, 0, stream>>>(w_qkv, wtq, DMODEL, NQKV);
        dim3 g2(DMODEL / 32, DMODEL / 32);
        transpose_bf16_kernel<<<g2, 256, 0, stream>>>(w_out, wto, DMODEL, DMODEL);
    }
    // 1) qkv = x @ w_qkv + b_qkv      (M=8192, N=6144, K=2048) -> bf16
    {
        dim3 grid(NQKV / 128, MROWS / 128);
        gemm_wmma_bf16<true><<<grid, 256, 0, stream>>>(xb, wtq, b_qkv, qkv,
                                                       MROWS, NQKV, DMODEL);
    }
    // 2) RoPE + head split + v transpose
    {
        const int total = BATCH * NHEADS * SEQ * 64;
        rope_split_kernel<<<total / 256, 256, 0, stream>>>(qkv, r_cos, r_sin, qb, kb, vtb);
    }
    // 3) causal flash attention (one wave per 16-query tile) -> bf16 o
    {
        const int tasks = BATCH * NHEADS * (SEQ / 16);   // 8192 waves
        attn_wmma_kernel<<<tasks / 8, 256, 0, stream>>>(qb, kb, vtb, ob);
    }
    // 4) out = o @ w_out + b_out      (M=8192, N=2048, K=2048) -> f32
    {
        dim3 grid(DMODEL / 128, MROWS / 128);
        gemm_wmma_bf16<false><<<grid, 256, 0, stream>>>(ob, wto, b_out, (float*)d_out,
                                                        MROWS, DMODEL, DMODEL);
    }
}
